// BDH_GPU_65326452572468
// MI455X (gfx1250) — matrix operand order repaired and checked by reading, not compiled, not run on hardware
//
#include <hip/hip_runtime.h>

// ---------------- problem constants ----------------
#define B_    4
#define T_    2048
#define D_    128
#define H_    4
#define DH_   32
#define NH_   1024
#define ND_   4096       // N = H*NH
#define VOC_  256
#define NTOK  (B_*T_)    // 8192

typedef unsigned short u16;
typedef __bf16 bf16x8  __attribute__((ext_vector_type(8)));
typedef __bf16 bf16x16 __attribute__((ext_vector_type(16)));
typedef float  f32x8   __attribute__((ext_vector_type(8)));
typedef int    v4i     __attribute__((vector_size(16)));

#define AS1 __attribute__((address_space(1)))
#define AS3 __attribute__((address_space(3)))

#if defined(__has_builtin)
#  if __has_builtin(__builtin_amdgcn_global_load_async_to_lds_b128)
#    define HAVE_ASYNC_LDS 1
#  endif
#endif

__device__ __forceinline__ void wait_async0() {
#if defined(__has_builtin) && __has_builtin(__builtin_amdgcn_s_wait_asynccnt)
  __builtin_amdgcn_s_wait_asynccnt(0);
#else
  asm volatile("s_wait_asynccnt 0x0" ::: "memory");
#endif
}

// ---------------- bf16 helpers (manual RNE, storage = u16) ----------------
__device__ __forceinline__ u16 f2bf(float f) {
  union { float f; unsigned u; } c; c.f = f;
  unsigned u = c.u + 0x7FFFu + ((c.u >> 16) & 1u);
  return (u16)(u >> 16);
}
__device__ __forceinline__ float bf2f(u16 h) {
  union { unsigned u; float f; } c; c.u = ((unsigned)h) << 16;
  return c.f;
}

// ---------------- WMMA fragment helpers ----------------
// A 16x32 bf16 (contiguous k, same as B): lane<16 -> row M=lane, K = 0..15
//                            lane>=16 -> row M=lane-16, K = 16..31
// per lane: two contiguous 16B chunks at kbase and kbase+8, kbase=(lane>=16)*16.
__device__ __forceinline__ bf16x16 frag_lohi(const u16* lo, const u16* hi) {
  bf16x8 a = *(const bf16x8*)lo;
  bf16x8 b = *(const bf16x8*)hi;
  bf16x16 r;
#pragma unroll
  for (int i = 0; i < 8; ++i) { r[i] = a[i]; r[i + 8] = b[i]; }
  return r;
}
// B 32x16 bf16: lane half -> K = khalf*16 + e (e = 0..15 contiguous), N = lane&15
__device__ __forceinline__ bf16x16 frag16(const u16* p) {
  return *(const bf16x16*)p;   // 32B, all call sites 32B-aligned
}
__device__ __forceinline__ f32x8 wmma_bf16(bf16x16 a, bf16x16 b, f32x8 c) {
  return __builtin_amdgcn_wmma_f32_16x16x32_bf16(false, a, false, b,
                                                 (short)0, c, false, false);
}

// ---------------- conversion kernels ----------------
__global__ void k_cvt(const float* __restrict__ s, u16* __restrict__ d, int n) {
  int i = blockIdx.x * blockDim.x + threadIdx.x;
  if (i < n) d[i] = f2bf(s[i]);
}
// decoder (H, DH, NH) -> bf16 (H, NH, DH)
__global__ void k_cvt_wT(const float* __restrict__ s, u16* __restrict__ d) {
  int i = blockIdx.x * blockDim.x + threadIdx.x;     // over H*DH*NH = 131072
  int h = i >> 15, r = i & 32767;
  int k = r >> 10, col = r & 1023;
  d[(((size_t)h << 10) + col) * DH_ + k] = f2bf(s[i]);
}
// encoder (ND, D) -> bf16 (D, ND)
__global__ void k_cvt_encT(const float* __restrict__ s, u16* __restrict__ d) {
  int i = blockIdx.x * blockDim.x + threadIdx.x;     // over ND*D = 524288
  int n = i >> 7, dd = i & 127;
  d[(size_t)dd * ND_ + n] = f2bf(s[i]);
}

// ---------------- embedding ----------------
__global__ void k_embed(const int* __restrict__ idx, const float* __restrict__ wte,
                        float* __restrict__ v) {
  int tok = blockIdx.x, d = threadIdx.x;
  v[(size_t)tok * D_ + d] = wte[(size_t)idx[tok] * D_ + d];
}

// ---------------- layernorm helpers ----------------
__device__ __forceinline__ void ln_stats(float x, float& mu, float& rstd) {
  __shared__ float red[128];
  int d = threadIdx.x;
  red[d] = x; __syncthreads();
#pragma unroll
  for (int s = 64; s > 0; s >>= 1) { if (d < s) red[d] += red[d + s]; __syncthreads(); }
  mu = red[0] * (1.0f / 128.0f); __syncthreads();
  float xc = x - mu;
  red[d] = xc * xc; __syncthreads();
#pragma unroll
  for (int s = 64; s > 0; s >>= 1) { if (d < s) red[d] += red[d + s]; __syncthreads(); }
  rstd = rsqrtf(red[0] * (1.0f / 128.0f) + 1e-5f); __syncthreads();
}

// vn = LN(v); store row-major bf16 and transposed (B, D, T) bf16 (for V-fragments)
__global__ void k_ln1(const float* __restrict__ v, const float* __restrict__ g,
                      const float* __restrict__ b, u16* __restrict__ vn,
                      u16* __restrict__ vnT) {
  int tok = blockIdx.x, d = threadIdx.x;
  float x = v[(size_t)tok * D_ + d];
  float mu, rstd; ln_stats(x, mu, rstd);
  float y = (x - mu) * rstd * g[d] + b[d];
  u16 h = f2bf(y);
  vn[(size_t)tok * D_ + d] = h;
  int bb = tok / T_, t = tok - bb * T_;
  vnT[((size_t)bb * D_ + d) * T_ + t] = h;
}

// v = v + LN(v); also emit bf16 copy of the new v
__global__ void k_ln2add(float* __restrict__ v, const float* __restrict__ g,
                         const float* __restrict__ b, u16* __restrict__ vbf) {
  int tok = blockIdx.x, d = threadIdx.x;
  float x = v[(size_t)tok * D_ + d];
  float mu, rstd; ln_stats(x, mu, rstd);
  float y = x + ((x - mu) * rstd * g[d] + b[d]);
  v[(size_t)tok * D_ + d] = y;
  vbf[(size_t)tok * D_ + d] = f2bf(y);
}

// ---------------- x = relu(vn_r @ Wx), xr = rope(x)  (one WMMA per tile, K=DH=32)
__global__ __launch_bounds__(256) void k_qkproj(const u16* __restrict__ vn,
                                                const u16* __restrict__ wxT,
                                                u16* __restrict__ xbf,
                                                u16* __restrict__ xrbf) {
  int w = threadIdx.x >> 5, lane = threadIdx.x & 31;
  int khalf = lane >> 4, nn = lane & 15;
  int bh = blockIdx.z, b = bh >> 2, h = bh & 3;
  int t0 = blockIdx.y * 16;
  int i0 = (blockIdx.x * 8 + w) * 16;

  const u16* ap = vn + ((size_t)(b * T_ + t0 + nn)) * D_ + h * DH_ + khalf * 16;
  bf16x16 A = frag_lohi(ap, ap + 8);
  bf16x16 Bf = frag16(wxT + ((size_t)h * NH_ + i0 + nn) * DH_ + khalf * 16);
  f32x8 c = {};
  c = wmma_bf16(A, Bf, c);

  int icol = i0 + nn;
  int kpair = icol >> 1;
  float freq = __expf(-(float)kpair * (9.210340371976184f / 512.0f)); // 10000^(-2k/1024)
#pragma unroll
  for (int e = 0; e < 8; ++e) {
    int t = t0 + e + khalf * 8;
    float x = fmaxf(c[e], 0.0f);
    float partner = __shfl_xor(x, 1, 32);   // rope pairs are adjacent lanes
    float sv, cv; __sincosf((float)t * freq, &sv, &cv);
    float xe = (icol & 1) ? partner : x;
    float xo = (icol & 1) ? x : partner;
    float rot = (icol & 1) ? (xe * sv + xo * cv) : (xe * cv - xo * sv);
    size_t o = ((size_t)bh * T_ + t) * NH_ + icol;
    xbf[o]  = f2bf(x);
    xrbf[o] = f2bf(rot);
  }
}

// ---------------- flash attention: S = xr@xr^T (K=1024), causal softmax, O = P@V
// Each wave owns 16 query rows; its 16x1024 Q block is staged once into LDS
// (async global->LDS if available), then read back as ds_load_b128 A-fragments.
__global__ __launch_bounds__(128) void k_attn(const u16* __restrict__ xr,
                                              const u16* __restrict__ vnT,
                                              u16* __restrict__ abf) {
  __shared__ __align__(16) u16 qtile[4][16 * NH_];  // 4 waves x 32 KB = 128 KB
  __shared__ __align__(16) u16 pbuf[4][16 * 32];    // per-wave P transpose, 4 KB
  int w = threadIdx.x >> 5, lane = threadIdx.x & 31;
  int khalf = lane >> 4, nn = lane & 15;
  int bh = blockIdx.y, b = bh >> 2, h = bh & 3;
  int t0 = blockIdx.x * 64 + w * 16;

  const u16* xq = xr + (size_t)bh * T_ * NH_;
  const u16* vb = vnT + ((size_t)b * D_ + h * DH_) * T_;

  // ---- stage this wave's Q block: 16 rows x 1024 bf16 = 2048 x 16B chunks ----
  u16* qs = qtile[w];
#ifdef HAVE_ASYNC_LDS
#pragma unroll
  for (int it = 0; it < 64; ++it) {
    int chunk = it * 32 + lane;          // 16B chunk index
    int row = chunk >> 7;                // 128 chunks per 2KB row
    int c8  = (chunk & 127) * 8;         // u16 offset within row
    __builtin_amdgcn_global_load_async_to_lds_b128(
        (AS1 v4i*)(xq + (size_t)(t0 + row) * NH_ + c8),
        (AS3 v4i*)(qs + row * NH_ + c8), 0, 0);
  }
  wait_async0();
#else
#pragma unroll
  for (int it = 0; it < 64; ++it) {
    int chunk = it * 32 + lane;
    int row = chunk >> 7;
    int c8  = (chunk & 127) * 8;
    *(bf16x8*)(qs + row * NH_ + c8) =
        *(const bf16x8*)(xq + (size_t)(t0 + row) * NH_ + c8);
  }
#endif

  float mrow[8], lrow[8];
  f32x8 o0 = {}, o1 = {};
#pragma unroll
  for (int e = 0; e < 8; ++e) { mrow[e] = -1e30f; lrow[e] = 0.0f; }

  for (int s0 = 0; s0 <= t0 + 15; s0 += 32) {       // 32 keys per step
    if (s0 + 32 <= t0 + 15) {                       // uniform per wave
      __builtin_prefetch(xq + (size_t)(s0 + 32 + nn) * NH_, 0, 1);
      __builtin_prefetch(xq + (size_t)(s0 + 48 + nn) * NH_, 0, 1);
    }
    f32x8 sa = {}, sb = {};
#pragma unroll 8
    for (int kk = 0; kk < NH_; kk += 32) {
      const u16* ap = qs + nn * NH_ + kk + khalf * 16;   // A from LDS
      bf16x16 A = frag_lohi(ap, ap + 8);
      bf16x16 B0 = frag16(xq + (size_t)(s0 + nn) * NH_ + kk + khalf * 16);
      bf16x16 B1 = frag16(xq + (size_t)(s0 + 16 + nn) * NH_ + kk + khalf * 16);
      sa = wmma_bf16(A, B0, sa);
      sb = wmma_bf16(A, B1, sb);
    }
    // causal mask + online softmax (row = e + 8*khalf, replicated across 16 lanes)
#pragma unroll
    for (int e = 0; e < 8; ++e) {
      int t = t0 + e + khalf * 8;
      if (s0 + nn > t)      sa[e] = -1e30f;
      if (s0 + 16 + nn > t) sb[e] = -1e30f;
      float mx = fmaxf(sa[e], sb[e]);
      mx = fmaxf(mx, __shfl_xor(mx, 1, 32));
      mx = fmaxf(mx, __shfl_xor(mx, 2, 32));
      mx = fmaxf(mx, __shfl_xor(mx, 4, 32));
      mx = fmaxf(mx, __shfl_xor(mx, 8, 32));
      float mnew = fmaxf(mrow[e], mx);
      float alpha = __expf(mrow[e] - mnew);
      float p0 = __expf(sa[e] - mnew);
      float p1 = __expf(sb[e] - mnew);
      float rs = p0 + p1;
      rs += __shfl_xor(rs, 1, 32);
      rs += __shfl_xor(rs, 2, 32);
      rs += __shfl_xor(rs, 4, 32);
      rs += __shfl_xor(rs, 8, 32);
      lrow[e] = lrow[e] * alpha + rs;
      mrow[e] = mnew;
      o0[e] *= alpha; o1[e] *= alpha;
      sa[e] = p0; sb[e] = p1;
    }
    // C-layout P -> A-layout P via per-wave LDS (same-wave DS ops are in-order)
    u16* pb = pbuf[w];
#pragma unroll
    for (int e = 0; e < 8; ++e) {
      int r = e + khalf * 8;
      pb[r * 32 + nn]      = f2bf(sa[e]);
      pb[r * 32 + 16 + nn] = f2bf(sb[e]);
    }
    const u16* pl = pb + nn * 32 + khalf * 16;
    bf16x16 P = frag_lohi(pl, pl + 8);
    // V fragments from vnT (contiguous over K = s)
    o0 = wmma_bf16(P, frag16(vb + (size_t)(0  + nn) * T_ + s0 + khalf * 16), o0);
    o1 = wmma_bf16(P, frag16(vb + (size_t)(16 + nn) * T_ + s0 + khalf * 16), o1);
  }
#pragma unroll
  for (int e = 0; e < 8; ++e) {
    int t = t0 + e + khalf * 8;
    float inv = 1.0f / lrow[e];
    abf[((size_t)bh * T_ + t) * DH_ + nn]      = f2bf(o0[e] * inv);
    abf[((size_t)bh * T_ + t) * DH_ + 16 + nn] = f2bf(o1[e] * inv);
  }
}

// ---------------- yi = relu(a @ Wy) * x ; scatter to y (B,T,N) ----------------
__global__ __launch_bounds__(256) void k_yproj(const u16* __restrict__ abf,
                                               const u16* __restrict__ wyT,
                                               const u16* __restrict__ xbf,
                                               u16* __restrict__ ybf) {
  int w = threadIdx.x >> 5, lane = threadIdx.x & 31;
  int khalf = lane >> 4, nn = lane & 15;
  int bh = blockIdx.z, b = bh >> 2, h = bh & 3;
  int t0 = blockIdx.y * 16;
  int i0 = (blockIdx.x * 8 + w) * 16;

  const u16* ap = abf + ((size_t)bh * T_ + t0 + nn) * DH_ + khalf * 16;
  bf16x16 A = frag_lohi(ap, ap + 8);
  bf16x16 Bf = frag16(wyT + ((size_t)h * NH_ + i0 + nn) * DH_ + khalf * 16);
  f32x8 c = {};
  c = wmma_bf16(A, Bf, c);

  int icol = i0 + nn;
#pragma unroll
  for (int e = 0; e < 8; ++e) {
    int t = t0 + e + khalf * 8;
    float xv = bf2f(xbf[((size_t)bh * T_ + t) * NH_ + icol]);
    float yv = fmaxf(c[e], 0.0f) * xv;
    ybf[((size_t)(b * T_ + t)) * ND_ + h * NH_ + icol] = f2bf(yv);
  }
}

// ---------------- v += y @ encoder  (M=8192, N=128, K=4096) ----------------
__global__ __launch_bounds__(256) void k_encgemm(const u16* __restrict__ ybf,
                                                 const u16* __restrict__ encT,
                                                 float* __restrict__ v) {
  int w = threadIdx.x >> 5, lane = threadIdx.x & 31;
  int khalf = lane >> 4, nn = lane & 15;
  int d0 = blockIdx.x * 16;
  int m0 = (blockIdx.y * 8 + w) * 16;
  f32x8 c = {};
#pragma unroll 8
  for (int kk = 0; kk < ND_; kk += 32) {
    const u16* ap = ybf + (size_t)(m0 + nn) * ND_ + kk + khalf * 16;
    bf16x16 A = frag_lohi(ap, ap + 8);
    bf16x16 Bf = frag16(encT + (size_t)(d0 + nn) * ND_ + kk + khalf * 16);
    c = wmma_bf16(A, Bf, c);
  }
#pragma unroll
  for (int e = 0; e < 8; ++e) {
    int row = m0 + e + khalf * 8;
    float* p = v + (size_t)row * D_ + d0 + nn;
    *p += c[e];
  }
}

// ---------------- logits = v @ readout^T (M=8192, N=256, K=128) ----------------
__global__ __launch_bounds__(256) void k_logits(const u16* __restrict__ vbf,
                                                const u16* __restrict__ robf,
                                                float* __restrict__ out) {
  int w = threadIdx.x >> 5, lane = threadIdx.x & 31;
  int khalf = lane >> 4, nn = lane & 15;
  int vo0 = blockIdx.x * 16;
  int m0 = (blockIdx.y * 8 + w) * 16;
  f32x8 c = {};
#pragma unroll
  for (int kk = 0; kk < D_; kk += 32) {
    const u16* ap = vbf + (size_t)(m0 + nn) * D_ + kk + khalf * 16;
    bf16x16 A = frag_lohi(ap, ap + 8);
    bf16x16 Bf = frag16(robf + (size_t)(vo0 + nn) * D_ + kk + khalf * 16);
    c = wmma_bf16(A, Bf, c);
  }
#pragma unroll
  for (int e = 0; e < 8; ++e) {
    int row = m0 + e + khalf * 8;
    out[(size_t)row * VOC_ + vo0 + nn] = c[e];
  }
}

// ---------------- host driver ----------------
extern "C" void kernel_launch(void* const* d_in, const int* in_sizes, int n_in,
                              void* d_out, int out_size, void* d_ws, size_t ws_size,
                              hipStream_t stream) {
  (void)in_sizes; (void)n_in; (void)out_size; (void)ws_size;
  const int*   idx  = (const int*)d_in[0];
  const float* wte  = (const float*)d_in[1];
  const float* enc  = (const float*)d_in[2];
  const float* decx = (const float*)d_in[3];
  const float* decy = (const float*)d_in[4];
  const float* l1g  = (const float*)d_in[5];
  const float* l1b  = (const float*)d_in[6];
  const float* l2g  = (const float*)d_in[7];
  const float* l2b  = (const float*)d_in[8];
  const float* rw   = (const float*)d_in[9];
  float* out = (float*)d_out;

  char* ws = (char*)d_ws;
  size_t off = 0;
  auto take = [&](size_t bytes) -> char* {
    char* p = ws + off;
    off += (bytes + 255) & ~(size_t)255;
    return p;
  };
  float* v    = (float*)take((size_t)NTOK * D_ * 4);          // 4 MB
  u16* vn     = (u16*)take((size_t)NTOK * D_ * 2);            // 2 MB
  u16* vnT    = (u16*)take((size_t)NTOK * D_ * 2);            // 2 MB
  u16* vbf    = (u16*)take((size_t)NTOK * D_ * 2);            // 2 MB
  u16* abf    = (u16*)take((size_t)B_ * H_ * T_ * DH_ * 2);   // 2 MB
  u16* xbf    = (u16*)take((size_t)B_ * H_ * T_ * NH_ * 2);   // 64 MB
  u16* xrbf   = (u16*)take((size_t)B_ * H_ * T_ * NH_ * 2);   // 64 MB
  u16* ybf    = (u16*)take((size_t)NTOK * ND_ * 2);           // 64 MB
  u16* wxT    = (u16*)take((size_t)H_ * NH_ * DH_ * 2);
  u16* wyT    = (u16*)take((size_t)H_ * NH_ * DH_ * 2);
  u16* encT   = (u16*)take((size_t)D_ * ND_ * 2);
  u16* robf   = (u16*)take((size_t)VOC_ * D_ * 2);

  // weight conversions (deterministic, re-done every call)
  k_cvt_wT  <<<(H_ * DH_ * NH_) / 256, 256, 0, stream>>>(decx, wxT);
  k_cvt_wT  <<<(H_ * DH_ * NH_) / 256, 256, 0, stream>>>(decy, wyT);
  k_cvt_encT<<<(ND_ * D_) / 256,      256, 0, stream>>>(enc, encT);
  k_cvt     <<<(VOC_ * D_) / 256,     256, 0, stream>>>(rw, robf, VOC_ * D_);

  k_embed<<<NTOK, 128, 0, stream>>>(idx, wte, v);

  for (int l = 0; l < 4; ++l) {
    k_ln1   <<<NTOK, 128, 0, stream>>>(v, l1g, l1b, vn, vnT);
    k_qkproj<<<dim3(NH_ / 128, T_ / 16, B_ * H_), 256, 0, stream>>>(vn, wxT, xbf, xrbf);
    k_attn  <<<dim3(T_ / 64, B_ * H_),            128, 0, stream>>>(xrbf, vnT, abf);
    k_yproj <<<dim3(NH_ / 128, T_ / 16, B_ * H_), 256, 0, stream>>>(abf, wyT, xbf, ybf);
    k_encgemm<<<dim3(D_ / 16, NTOK / 128),        256, 0, stream>>>(ybf, encT, v);
    k_ln2add<<<NTOK, 128, 0, stream>>>(v, l2g, l2b, vbf);
  }
  k_logits<<<dim3(VOC_ / 16, NTOK / 128), 256, 0, stream>>>(vbf, robf, out);
}
